// MultiHeadAttention_50405736185978
// MI455X (gfx1250) — matrix operand order repaired and checked by reading, not compiled, not run on hardware
//
#include <hip/hip_runtime.h>
#include <hip/hip_bf16.h>

#define BB 2
#define TT 2048
#define DD 1024
#define HH 16
#define DHD 64
#define MMR (BB * TT)

typedef __bf16 v16bf __attribute__((ext_vector_type(16)));
typedef float v8f __attribute__((ext_vector_type(8)));
typedef unsigned short u16x4 __attribute__((ext_vector_type(4)));
typedef unsigned short u16x8 __attribute__((ext_vector_type(8)));
typedef unsigned short u16x16 __attribute__((ext_vector_type(16)));

__device__ __forceinline__ unsigned short f2bf(float f) {
    unsigned int u = __float_as_uint(f);
    u += 0x7FFFu + ((u >> 16) & 1u);
    return (unsigned short)(u >> 16);
}

// A-operand fragment (16x32, 16-bit): lane half h -> K = {h*8..h*8+7, 16+h*8..+7}
__device__ __forceinline__ v16bf load_frag_a(const unsigned short* rowbase, int half) {
    u16x8 lo = *(const u16x8*)(rowbase + half * 8);
    u16x8 hi = *(const u16x8*)(rowbase + 16 + half * 8);
    union { u16x16 u; v16bf v; } f;
#pragma unroll
    for (int i = 0; i < 8; ++i) { f.u[i] = lo[i]; f.u[8 + i] = hi[i]; }
    return f.v;
}

// B-operand fragment (32x16, 16-bit): lane half h -> K = {h*8..h*8+7, 16+h*8..+7}
__device__ __forceinline__ v16bf load_frag_b(const unsigned short* rowbase, int half) {
    u16x8 lo = *(const u16x8*)(rowbase + half * 8);
    u16x8 hi = *(const u16x8*)(rowbase + 16 + half * 8);
    union { u16x16 u; v16bf v; } f;
#pragma unroll
    for (int i = 0; i < 8; ++i) { f.u[i] = lo[i]; f.u[8 + i] = hi[i]; }
    return f.v;
}

__device__ __forceinline__ v8f wmma_bf16(v16bf a, v16bf b, v8f c) {
    return __builtin_amdgcn_wmma_f32_16x16x32_bf16(false, a, false, b, (short)0, c,
                                                   false, false);
}

// ---------------------------------------------------------------------------
// fp32 -> bf16 bulk convert (activations)
// ---------------------------------------------------------------------------
__global__ __launch_bounds__(256) void cvt_bf16(const float* __restrict__ in,
                                                unsigned short* __restrict__ out,
                                                int n) {
    int i = (blockIdx.x * 256 + threadIdx.x) * 4;
    if (i + 3 < n) {
        float4 v = *(const float4*)(in + i);
        u16x4 o;
        o[0] = f2bf(v.x); o[1] = f2bf(v.y); o[2] = f2bf(v.z); o[3] = f2bf(v.w);
        *(u16x4*)(out + i) = o;
    }
}

// ---------------------------------------------------------------------------
// W [K][N] fp32 -> Wt [N][K] bf16 (tiled transpose through LDS)
// ---------------------------------------------------------------------------
__global__ __launch_bounds__(256) void cvt_wt(const float* __restrict__ W,
                                              unsigned short* __restrict__ Wt) {
    __shared__ unsigned short tile[32][33];
    const int k0 = blockIdx.y * 32, n0 = blockIdx.x * 32;
    const int tx = threadIdx.x & 31, ty = threadIdx.x >> 5;   // 32 x 8
#pragma unroll
    for (int i = 0; i < 4; ++i) {
        int kk = ty + i * 8;
        tile[kk][tx] = f2bf(W[(size_t)(k0 + kk) * DD + n0 + tx]);
    }
    __syncthreads();
#pragma unroll
    for (int i = 0; i < 4; ++i) {
        int nn = ty + i * 8;
        Wt[(size_t)(n0 + nn) * DD + k0 + tx] = tile[tx][nn];
    }
}

// ---------------------------------------------------------------------------
// WMMA GEMM, all-bf16 operands, async-to-LDS double-buffered staging.
// Out = Xb @ Wt^T + bias  (Wt stored [N][K] so both tiles copy contiguously).
// Block tile 64x64, BLOCK_K=64 (2 WMMA k-steps per stage), 8 waves.
// MODE 0: bf16 out, head-split [B][H][T][64]. MODE 1: bf16 out, [B][H][64][T].
// MODE 2: fp32 plain out.
// ---------------------------------------------------------------------------
template <int MODE>
__global__ __launch_bounds__(256) void gemm_wmma(const unsigned short* __restrict__ Xb,
                                                 const unsigned short* __restrict__ Wt,
                                                 const float* __restrict__ bias,
                                                 void* __restrict__ Out,
                                                 int M, int N, int K) {
    __shared__ unsigned short sA[2][64 * 64];   // [m][k] bf16
    __shared__ unsigned short sB[2][64 * 64];   // [n][k] bf16
    const int tid = threadIdx.x;
    const int m0 = blockIdx.y * 64, n0 = blockIdx.x * 64;
    const int wid = tid >> 5, lane = tid & 31;
    const int lane15 = lane & 15, half = lane >> 4;
    const int tn = wid & 3, tm = (wid >> 2) * 2;

    const unsigned sA_base = (unsigned)(size_t)(&sA[0][0]);
    const unsigned sB_base = (unsigned)(size_t)(&sB[0][0]);

    // Each thread async-copies 2x16B of the A tile and 2x16B of the B tile.
    auto issue_tile = [&](int buf, int k0) {
#pragma unroll
        for (int i = 0; i < 2; ++i) {
            int chunk = tid + i * 256;       // 0..511 -> 8KB tile
            int mm = chunk >> 3;             // row (128B per row)
            int kb = (chunk & 7) * 16;       // byte offset within row
            unsigned long long ga =
                (unsigned long long)(Xb + (size_t)(m0 + mm) * K + k0 + (kb >> 1));
            unsigned la = sA_base + (unsigned)(buf * 8192 + mm * 128 + kb);
            asm volatile("global_load_async_to_lds_b128 %0, %1, off"
                         :: "v"(la), "v"(ga) : "memory");
            unsigned long long gb =
                (unsigned long long)(Wt + (size_t)(n0 + mm) * K + k0 + (kb >> 1));
            unsigned lb = sB_base + (unsigned)(buf * 8192 + mm * 128 + kb);
            asm volatile("global_load_async_to_lds_b128 %0, %1, off"
                         :: "v"(lb), "v"(gb) : "memory");
        }
    };

    v8f acc0 = {}, acc1 = {};
    const int NT = K / 64;
    issue_tile(0, 0);
    for (int t = 0; t < NT; ++t) {
        const int buf = t & 1;
        if (t + 1 < NT) {
            issue_tile(buf ^ 1, (t + 1) * 64);
            asm volatile("s_wait_asynccnt 0x4" ::: "memory");   // tile t landed
        } else {
            asm volatile("s_wait_asynccnt 0x0" ::: "memory");
        }
        __syncthreads();
        const unsigned short* A = &sA[buf][0];
        const unsigned short* Bp = &sB[buf][0];
#pragma unroll
        for (int ks = 0; ks < 2; ++ks) {
            v16bf bf = load_frag_b(Bp + (tn * 16 + lane15) * 64 + ks * 32, half);
            v16bf a0 = load_frag_a(A + ((tm + 0) * 16 + lane15) * 64 + ks * 32, half);
            v16bf a1 = load_frag_a(A + ((tm + 1) * 16 + lane15) * 64 + ks * 32, half);
            acc0 = wmma_bf16(a0, bf, acc0);
            acc1 = wmma_bf16(a1, bf, acc1);
        }
        __syncthreads();   // all waves done reading buf before it is re-filled
    }

    const int col = n0 + tn * 16 + lane15;
    const float bv = bias[col];
#pragma unroll
    for (int t = 0; t < 2; ++t) {
        v8f acc = t ? acc1 : acc0;
#pragma unroll
        for (int r = 0; r < 8; ++r) {
            int row = m0 + (tm + t) * 16 + half * 8 + r;
            float val = acc[r] + bv;
            if (MODE == 2) {
                ((float*)Out)[(size_t)row * N + col] = val;
            } else {
                int b = row >> 11;            // / TT
                int tq = row & (TT - 1);
                int h = col >> 6;             // / DHD
                int dh = col & (DHD - 1);
                size_t off;
                if (MODE == 0)
                    off = (((size_t)(b * HH + h) * TT) + tq) * DHD + dh;
                else
                    off = (((size_t)(b * HH + h) * DHD) + dh) * TT + tq;
                ((unsigned short*)Out)[off] = f2bf(val);
            }
        }
    }
}

// ---------------------------------------------------------------------------
// Flash attention, causal. One wave per 16-row query tile; 64 keys/step
// (amortizes the online-softmax bookkeeping: one max-butterfly, one alpha
// rescale and one mrow/lsum update per 64 keys). Row-sum kept as per-lane
// partial, single butterfly at the end. Softmax in exp2 domain.
// With k0 a multiple of 64 and k0 < q0+16 <= T, max key row = k0+63 <= T-1,
// so all K/V reads are in-bounds without clamping.
// ---------------------------------------------------------------------------
struct AttnState {
    v8f o[4];
    float mrow[8];
    float lsum[8];
};

template <bool MASKED>
__device__ __forceinline__ void attn_step64(const unsigned short* __restrict__ Kh,
                                            const unsigned short* __restrict__ Vh,
                                            unsigned short* __restrict__ myP,
                                            v16bf qf0, v16bf qf1, AttnState& st,
                                            int k0, int q0, int lane15, int half) {
    v8f s[4];
#pragma unroll
    for (int j = 0; j < 4; ++j) {
        const size_t kr = (size_t)(k0 + j * 16 + lane15);
        v16bf b0 = load_frag_b(Kh + kr * DHD + 0, half);
        v16bf b1 = load_frag_b(Kh + kr * DHD + 32, half);
        v8f z = {};
        z = wmma_bf16(qf0, b0, z);
        s[j] = wmma_bf16(qf1, b1, z);
    }
    const float C = 0.125f * 1.44269504088896f;        // 1/sqrt(64) * log2(e)
#pragma unroll
    for (int r = 0; r < 8; ++r) {
        const int qrow = q0 + half * 8 + r;
        float v0 = s[0][r] * C;
        float v1 = s[1][r] * C;
        float v2 = s[2][r] * C;
        float v3 = s[3][r] * C;
        if (MASKED) {
            if (k0 + lane15 > qrow)      v0 = -1e30f;
            if (k0 + 16 + lane15 > qrow) v1 = -1e30f;
            if (k0 + 32 + lane15 > qrow) v2 = -1e30f;
            if (k0 + 48 + lane15 > qrow) v3 = -1e30f;
        }
        float mx = fmaxf(fmaxf(v0, v1), fmaxf(v2, v3));
        mx = fmaxf(mx, __shfl_xor(mx, 1, 32));
        mx = fmaxf(mx, __shfl_xor(mx, 2, 32));
        mx = fmaxf(mx, __shfl_xor(mx, 4, 32));
        mx = fmaxf(mx, __shfl_xor(mx, 8, 32));
        mx = fmaxf(mx, st.mrow[r]);
        float alpha = exp2f(st.mrow[r] - mx);
        float e0 = exp2f(v0 - mx);
        float e1 = exp2f(v1 - mx);
        float e2 = exp2f(v2 - mx);
        float e3 = exp2f(v3 - mx);
        st.lsum[r] = st.lsum[r] * alpha + ((e0 + e1) + (e2 + e3));
        st.mrow[r] = mx;
        st.o[0][r] *= alpha; st.o[1][r] *= alpha;
        st.o[2][r] *= alpha; st.o[3][r] *= alpha;
        const int lm = half * 8 + r;                   // C-layout -> LDS P tile
        myP[lm * 64 + lane15]      = f2bf(e0);
        myP[lm * 64 + 16 + lane15] = f2bf(e1);
        myP[lm * 64 + 32 + lane15] = f2bf(e2);
        myP[lm * 64 + 48 + lane15] = f2bf(e3);
    }
    asm volatile("s_wait_dscnt 0x0" ::: "memory");     // wave-local LDS ordering
    v16bf pf0 = load_frag_a(myP + lane15 * 64 + 0, half);
    v16bf pf1 = load_frag_a(myP + lane15 * 64 + 32, half);
#pragma unroll
    for (int j = 0; j < 4; ++j) {
        const unsigned short* vrow = Vh + (size_t)(j * 16 + lane15) * TT + k0;
        v16bf vb0 = load_frag_b(vrow + 0, half);
        v16bf vb1 = load_frag_b(vrow + 32, half);
        st.o[j] = wmma_bf16(pf0, vb0, st.o[j]);
        st.o[j] = wmma_bf16(pf1, vb1, st.o[j]);
    }
    asm volatile("s_wait_dscnt 0x0" ::: "memory");     // reads done before next writes
}

__global__ __launch_bounds__(128) void attn_flash(const unsigned short* __restrict__ Qb,
                                                  const unsigned short* __restrict__ Kb,
                                                  const unsigned short* __restrict__ Vt,
                                                  unsigned short* __restrict__ Ob) {
    __shared__ unsigned short sP[4][16 * 64];
    const int wid = threadIdx.x >> 5, lane = threadIdx.x & 31;
    const int lane15 = lane & 15, half = lane >> 4;
    const int bh = blockIdx.y;                  // b*H + h
    const int q0 = blockIdx.x * 64 + wid * 16;
    const unsigned short* Qh = Qb + (size_t)bh * TT * DHD;
    const unsigned short* Kh = Kb + (size_t)bh * TT * DHD;
    const unsigned short* Vh = Vt + (size_t)bh * DHD * TT;
    unsigned short* myP = sP[wid];

    v16bf qf0 = load_frag_a(Qh + (size_t)(q0 + lane15) * DHD + 0, half);
    v16bf qf1 = load_frag_a(Qh + (size_t)(q0 + lane15) * DHD + 32, half);

    AttnState st;
#pragma unroll
    for (int j = 0; j < 4; ++j) st.o[j] = v8f{};
#pragma unroll
    for (int r = 0; r < 8; ++r) { st.mrow[r] = -1e30f; st.lsum[r] = 0.0f; }

    const int kend = q0 + 16;                                    // causal extent
    const int kunm = (q0 >= 64) ? ((((q0 - 63) >> 6) + 1) << 6) : 0;
    int k0 = 0;
    for (; k0 < kunm; k0 += 64)
        attn_step64<false>(Kh, Vh, myP, qf0, qf1, st, k0, q0, lane15, half);
    for (; k0 < kend; k0 += 64)
        attn_step64<true>(Kh, Vh, myP, qf0, qf1, st, k0, q0, lane15, half);

    const int b = bh >> 4, h = bh & 15;
#pragma unroll
    for (int r = 0; r < 8; ++r) {
        float s = st.lsum[r];
        s += __shfl_xor(s, 1, 32);
        s += __shfl_xor(s, 2, 32);
        s += __shfl_xor(s, 4, 32);
        s += __shfl_xor(s, 8, 32);
        float inv = 1.0f / s;
        size_t row = (size_t)b * TT + q0 + half * 8 + r;
        size_t base = row * DD + (size_t)h * DHD + lane15;
        Ob[base + 0]  = f2bf(st.o[0][r] * inv);
        Ob[base + 16] = f2bf(st.o[1][r] * inv);
        Ob[base + 32] = f2bf(st.o[2][r] * inv);
        Ob[base + 48] = f2bf(st.o[3][r] * inv);
    }
}

extern "C" void kernel_launch(void* const* d_in, const int* in_sizes, int n_in,
                              void* d_out, int out_size, void* d_ws, size_t ws_size,
                              hipStream_t stream) {
    const float* query = (const float*)d_in[0];
    const float* key   = (const float*)d_in[1];
    const float* value = (const float*)d_in[2];
    const float* Wq = (const float*)d_in[3];
    const float* bq = (const float*)d_in[4];
    const float* Wk = (const float*)d_in[5];
    const float* bk = (const float*)d_in[6];
    const float* Wv = (const float*)d_in[7];
    const float* bv = (const float*)d_in[8];
    const float* Wo = (const float*)d_in[9];
    const float* bo = (const float*)d_in[10];
    // d_in[11] = mask: causal by construction, applied analytically in-kernel.

    const size_t segX = (size_t)MMR * DD + 128;         // bf16 elems (+pad)
    const size_t segW = (size_t)DD * DD + 128;
    unsigned short* p  = (unsigned short*)d_ws;
    unsigned short* Xq = p; p += segX;
    unsigned short* Xk = p; p += segX;
    unsigned short* Xv = p; p += segX;
    unsigned short* Wt = p; p += segW;                  // reused per projection
    unsigned short* Qb = p; p += segX;
    unsigned short* Kb = p; p += segX;
    unsigned short* Vt = p; p += segX;
    unsigned short* Ob = p; p += segX;

    const int nElemX = MMR * DD;
    dim3 cblk(256), cgrd(nElemX / 1024);
    cvt_bf16<<<cgrd, cblk, 0, stream>>>(query, Xq, nElemX);
    cvt_bf16<<<cgrd, cblk, 0, stream>>>(key,   Xk, nElemX);
    cvt_bf16<<<cgrd, cblk, 0, stream>>>(value, Xv, nElemX);

    dim3 tgrd(DD / 32, DD / 32), tblk(256);
    dim3 ggrd(DD / 64, MMR / 64), gblk(256);

    cvt_wt<<<tgrd, tblk, 0, stream>>>(Wq, Wt);
    gemm_wmma<0><<<ggrd, gblk, 0, stream>>>(Xq, Wt, bq, Qb, MMR, DD, DD);
    cvt_wt<<<tgrd, tblk, 0, stream>>>(Wk, Wt);
    gemm_wmma<0><<<ggrd, gblk, 0, stream>>>(Xk, Wt, bk, Kb, MMR, DD, DD);
    cvt_wt<<<tgrd, tblk, 0, stream>>>(Wv, Wt);
    gemm_wmma<1><<<ggrd, gblk, 0, stream>>>(Xv, Wt, bv, Vt, MMR, DD, DD);

    attn_flash<<<dim3(TT / 64, BB * HH), dim3(128), 0, stream>>>(Qb, Kb, Vt, Ob);

    cvt_wt<<<tgrd, tblk, 0, stream>>>(Wo, Wt);
    gemm_wmma<2><<<ggrd, gblk, 0, stream>>>(Ob, Wt, bo, d_out, MMR, DD, DD);
}
